// LunaHalfBlock_85341000171975
// MI455X (gfx1250) — compile-verified, matrix operand order checked by reading, not run on hardware
//
#include <hip/hip_runtime.h>
#include <math.h>

// ---------------- Problem constants ----------------
#define BATCH     8
#define NUM_HEADS 8
#define HEAD_DIM  64
#define HW        4096      // 64*64 spatial tokens
#define KTOK      256       // aux tokens
#define DMODEL    512

typedef __attribute__((ext_vector_type(16))) __bf16 v16bf;
typedef __attribute__((ext_vector_type(8)))  __bf16 v8bf;
typedef __attribute__((ext_vector_type(8)))  float  v8f;

union FragBF {
    v16bf  v;
    v8bf   h[2];
    __bf16 e[16];
};

// D = A(16x32 bf16) x B(32x16 bf16) + C, fp32 accum
__device__ inline v8f wmma_bf16(v16bf a, v16bf b, v8f c) {
    return __builtin_amdgcn_wmma_f32_16x16x32_bf16(
        /*neg_a=*/false, a, /*neg_b=*/false, b,
        /*c_mod=*/(short)0, c, /*reuse_a=*/false, /*reuse_b=*/false);
}

// Load a 16x32 (or 32x16) bf16 fragment for one lane.
// Per the CDNA5 16-bit A/B layout: lane-half selects K in
// {kc+half*8 .. +7} U {kc+16+half*8 .. +7}; both are contiguous 16B chunks.
__device__ inline v16bf load_frag(const __bf16* __restrict__ row, int kc, int half) {
    FragBF f;
    const int o = kc + half * 8;
    f.h[0] = *reinterpret_cast<const v8bf*>(row + o);
    f.h[1] = *reinterpret_cast<const v8bf*>(row + o + 16);
    return f.v;
}

// ---------------- hidden (b,c,h,w) -> hidbf (b, n=hw, c) bf16, LDS transpose ----------------
__global__ void transpose_hidden_kernel(const float* __restrict__ hidden,
                                        __bf16* __restrict__ hidbf) {
    __shared__ float tile[32][33];
    const int b  = blockIdx.z;
    const int nb = blockIdx.x * 32;
    const int cb = blockIdx.y * 32;
    const int tx = threadIdx.x;   // 32
    const int ty = threadIdx.y;   // 8
#pragma unroll
    for (int i = 0; i < 4; ++i) {
        int c = cb + ty + i * 8;
        tile[ty + i * 8][tx] = hidden[((size_t)b * DMODEL + c) * HW + nb + tx];
    }
    __syncthreads();
#pragma unroll
    for (int i = 0; i < 4; ++i) {
        int n = nb + ty + i * 8;
        hidbf[((size_t)b * HW + n) * DMODEL + cb + tx] = (__bf16)tile[tx][ty + i * 8];
    }
}

// ---------------- W (in,out) fp32 -> Wt (out,in) bf16 ----------------
__global__ void convT512_kernel(const float* __restrict__ W, __bf16* __restrict__ Wt) {
    int idx = blockIdx.x * 256 + threadIdx.x;   // 512*512 elements
    int i = idx >> 9, j = idx & 511;            // i = in, j = out
    Wt[(size_t)j * DMODEL + i] = (__bf16)W[idx];
}

// ---------------- fp32 -> bf16 copy ----------------
__global__ void conv_bf16_kernel(const float* __restrict__ src, __bf16* __restrict__ dst, int n) {
    int idx = blockIdx.x * 256 + threadIdx.x;
    if (idx < n) dst[idx] = (__bf16)src[idx];
}

// ---------------- generic M x 512 x 512 bf16 WMMA GEMM ----------------
// A: [M,512] bf16 row-major.  Bt: [512,512] bf16, row n holds column n of B.
// 8 waves/block; wave w computes rows [16*blockIdx.x,+16) x cols [64w,+64).
// mode 0: K-proj  -> out_bf = Kbf  [b][h][n][dq]
// mode 1: V-proj  -> out_bf = Vt   [b][h][dv][n]
// mode 2: Q-proj  -> out_bf = Qbf  [b][h][qk][dq], scaled by `scale`
// mode 3: O-proj  -> out_f[m][c] = acc + bias + resid[m][c]
__global__ void gemm512_bf16(const __bf16* __restrict__ A,
                             const __bf16* __restrict__ Bt,
                             const float*  __restrict__ bias,
                             const float*  __restrict__ resid,
                             __bf16* __restrict__ out_bf,
                             float*  __restrict__ out_f,
                             int mode, float scale) {
    const int lane  = threadIdx.x & 31;
    const int wave  = threadIdx.x >> 5;
    const int half  = lane >> 4;
    const int l16   = lane & 15;
    const int mtile = blockIdx.x;
    const int mrow  = mtile * 16 + l16;

    const __bf16* arow = A + (size_t)mrow * DMODEL;

    v8f acc[4] = {};
    for (int kc = 0; kc < DMODEL; kc += 32) {
        if (kc + 32 < DMODEL) __builtin_prefetch(arow + kc + 32, 0, 0);
        v16bf a = load_frag(arow, kc, half);
#pragma unroll
        for (int j = 0; j < 4; ++j) {
            const __bf16* brow = Bt + (size_t)(wave * 64 + j * 16 + l16) * DMODEL;
            v16bf bfrag = load_frag(brow, kc, half);
            acc[j] = wmma_bf16(a, bfrag, acc[j]);
        }
    }

#pragma unroll
    for (int j = 0; j < 4; ++j) {
        const int ccol = wave * 64 + j * 16 + l16;
        const float bval = bias ? bias[ccol] : 0.0f;
#pragma unroll
        for (int r = 0; r < 8; ++r) {
            const int m = mtile * 16 + r + half * 8;
            float v = acc[j][r] + bval;
            if (mode == 0) {            // K: [b][h][n][dq]
                int b = m >> 12, n = m & 4095, hh = ccol >> 6, dq = ccol & 63;
                out_bf[((size_t)(b * NUM_HEADS + hh) * HW + n) * HEAD_DIM + dq] = (__bf16)v;
            } else if (mode == 1) {     // Vt: [b][h][dv][n]
                int b = m >> 12, n = m & 4095, hh = ccol >> 6, dv = ccol & 63;
                out_bf[((size_t)(b * NUM_HEADS + hh) * HEAD_DIM + dv) * HW + n] = (__bf16)v;
            } else if (mode == 2) {     // Q (pre-scaled): [b][h][qk][dq]
                int b = m >> 8, qk = m & 255, hh = ccol >> 6, dq = ccol & 63;
                out_bf[((size_t)(b * NUM_HEADS + hh) * KTOK + qk) * HEAD_DIM + dq] =
                    (__bf16)(v * scale);
            } else {                    // O-proj + residual -> fp32
                out_f[(size_t)m * DMODEL + ccol] = v + resid[(size_t)m * DMODEL + ccol];
            }
        }
    }
}

// ---------------- attention: scores -> softmax (LDS strip) -> attn@V ----------------
// grid (KTOK/16, H, B), 256 threads (8 waves).
// Dynamic LDS: 16x4096 fp32 score strip (256 KB, fits CDNA5's 320 KB WGP LDS)
//            + 16x64 fp32 output reduce + softmax scratch.
__global__ void attention_kernel(const __bf16* __restrict__ Qbf,
                                 const __bf16* __restrict__ Kbf,
                                 const __bf16* __restrict__ Vt,
                                 __bf16* __restrict__ ctxbf,
                                 float* __restrict__ attn_out) {
    extern __shared__ float smem[];
    float* s_scores = smem;                       // 16*4096
    float* s_out    = smem + 16 * HW;             // 16*64
    float* s_red    = s_out + 16 * 64;            // 16*16
    float* s_stat   = s_red + 256;                // 16

    const int qtile = blockIdx.x;
    const int hh    = blockIdx.y;
    const int b     = blockIdx.z;
    const int tid   = threadIdx.x;
    const int lane  = tid & 31;
    const int wave  = tid >> 5;
    const int half  = lane >> 4;
    const int l16   = lane & 15;
    const size_t bh = (size_t)b * NUM_HEADS + hh;

    // ---- scores tile strip: Q(16x64, already scaled) @ K^T over this wave's 512 keys ----
    const __bf16* qrow = Qbf + (bh * KTOK + qtile * 16 + l16) * HEAD_DIM;
    v16bf a0 = load_frag(qrow, 0, half);
    v16bf a1 = load_frag(qrow, 32, half);

    for (int t = 0; t < 32; ++t) {
        const int nb = wave * 512 + t * 16;
        const __bf16* krow = Kbf + (bh * HW + nb + l16) * HEAD_DIM;
        v16bf b0 = load_frag(krow, 0, half);
        v16bf b1 = load_frag(krow, 32, half);
        v8f sc = {};
        sc = wmma_bf16(a0, b0, sc);
        sc = wmma_bf16(a1, b1, sc);
#pragma unroll
        for (int r = 0; r < 8; ++r)
            s_scores[(r + half * 8) * HW + nb + l16] = sc[r];
    }
    __syncthreads();

    // ---- fp32 softmax over each of the 16 rows (16 threads per row) ----
    const int row = tid >> 4;
    const int sub = tid & 15;
    float pmax = -3.4e38f;
    for (int c = sub; c < HW; c += 16) pmax = fmaxf(pmax, s_scores[row * HW + c]);
    s_red[row * 16 + sub] = pmax;
    __syncthreads();
    if (sub == 0) {
        float mm = s_red[row * 16];
        for (int i = 1; i < 16; ++i) mm = fmaxf(mm, s_red[row * 16 + i]);
        s_stat[row] = mm;
    }
    __syncthreads();
    const float rmax = s_stat[row];
    float lsum = 0.0f;
    for (int c = sub; c < HW; c += 16) {
        float e = __expf(s_scores[row * HW + c] - rmax);
        s_scores[row * HW + c] = e;
        lsum += e;
    }
    __syncthreads();
    s_red[row * 16 + sub] = lsum;
    __syncthreads();
    if (sub == 0) {
        float ss = 0.0f;
        for (int i = 0; i < 16; ++i) ss += s_red[row * 16 + i];
        s_stat[row] = ss;
    }
    __syncthreads();
    const float inv = 1.0f / s_stat[row];
    float* attn_row = attn_out + (bh * KTOK + qtile * 16 + row) * (size_t)HW;
    for (int c = sub; c < HW; c += 16) {
        float v = s_scores[row * HW + c] * inv;
        s_scores[row * HW + c] = v;   // keep normalized attn for attn@V
        attn_row[c] = v;              // attn is a kernel output
    }
    for (int e = tid; e < 16 * 64; e += 256) s_out[e] = 0.0f;
    __syncthreads();

    // ---- out = attn @ V over this wave's 512 keys, WMMA bf16 ----
    v8f oacc[4] = {};
    for (int t = 0; t < 16; ++t) {
        const int keybase = wave * 512 + t * 32;
        FragBF af;
#pragma unroll
        for (int i = 0; i < 16; ++i) {
            int kk = i + half * 8 + (i >= 8 ? 8 : 0);
            af.e[i] = (__bf16)s_scores[l16 * HW + keybase + kk];
        }
#pragma unroll
        for (int j = 0; j < 4; ++j) {
            const __bf16* vrow = Vt + (bh * HEAD_DIM + j * 16 + l16) * (size_t)HW + keybase;
            v16bf bfrag = load_frag(vrow, 0, half);
            oacc[j] = wmma_bf16(af.v, bfrag, oacc[j]);
        }
    }
#pragma unroll
    for (int j = 0; j < 4; ++j)
#pragma unroll
        for (int r = 0; r < 8; ++r)
            atomicAdd(&s_out[(r + half * 8) * 64 + j * 16 + l16], oacc[j][r]);
    __syncthreads();

    // ---- ctx (b, K, hidden) bf16, for the O-projection GEMM ----
    for (int e = tid; e < 16 * 64; e += 256) {
        int m = e >> 6, dv = e & 63;
        ctxbf[((size_t)b * KTOK + qtile * 16 + m) * DMODEL + hh * 64 + dv] =
            (__bf16)s_out[e];
    }
}

// ---------------- fp32 LayerNorm over 512, one block per row ----------------
__global__ void layernorm_kernel(const float* __restrict__ x,
                                 const float* __restrict__ g,
                                 const float* __restrict__ beta,
                                 float* __restrict__ out) {
    __shared__ float s_red[256];
    const int rowm = blockIdx.x;
    const int tid  = threadIdx.x;
    const float* xr = x + (size_t)rowm * DMODEL;
    const float v0 = xr[tid], v1 = xr[tid + 256];
    s_red[tid] = v0 + v1;
    __syncthreads();
    for (int s = 128; s > 0; s >>= 1) {
        if (tid < s) s_red[tid] += s_red[tid + s];
        __syncthreads();
    }
    const float mu = s_red[0] * (1.0f / DMODEL);
    __syncthreads();
    const float d0 = v0 - mu, d1 = v1 - mu;
    s_red[tid] = d0 * d0 + d1 * d1;
    __syncthreads();
    for (int s = 128; s > 0; s >>= 1) {
        if (tid < s) s_red[tid] += s_red[tid + s];
        __syncthreads();
    }
    const float inv = rsqrtf(s_red[0] * (1.0f / DMODEL) + 1e-5f);
    float* orow = out + (size_t)rowm * DMODEL;
    orow[tid]       = d0 * inv * g[tid] + beta[tid];
    orow[tid + 256] = d1 * inv * g[tid + 256] + beta[tid + 256];
}

// ---------------- host launch ----------------
extern "C" void kernel_launch(void* const* d_in, const int* in_sizes, int n_in,
                              void* d_out, int out_size, void* d_ws, size_t ws_size,
                              hipStream_t stream) {
    const float* hidden = (const float*)d_in[0];
    const float* aux    = (const float*)d_in[1];
    const float* Wq     = (const float*)d_in[2];
    const float* bq     = (const float*)d_in[3];
    const float* Wk     = (const float*)d_in[4];
    const float* bk     = (const float*)d_in[5];
    const float* Wv     = (const float*)d_in[6];
    const float* bv     = (const float*)d_in[7];
    const float* Wo     = (const float*)d_in[8];
    const float* bo     = (const float*)d_in[9];
    const float* ln_g   = (const float*)d_in[10];
    const float* ln_b   = (const float*)d_in[11];
    (void)in_sizes; (void)n_in; (void)out_size; (void)ws_size;

    char* ws = (char*)d_ws;
    size_t off = 0;
    auto alloc = [&](size_t bytes) -> void* {
        void* p = ws + off;
        off = (off + bytes + 255) & ~(size_t)255;
        return p;
    };
    __bf16* hidbf = (__bf16*)alloc((size_t)BATCH * HW * DMODEL * 2);
    __bf16* auxbf = (__bf16*)alloc((size_t)BATCH * KTOK * DMODEL * 2);
    __bf16* WqT   = (__bf16*)alloc((size_t)DMODEL * DMODEL * 2);
    __bf16* WkT   = (__bf16*)alloc((size_t)DMODEL * DMODEL * 2);
    __bf16* WvT   = (__bf16*)alloc((size_t)DMODEL * DMODEL * 2);
    __bf16* WoT   = (__bf16*)alloc((size_t)DMODEL * DMODEL * 2);
    __bf16* Qbf   = (__bf16*)alloc((size_t)BATCH * NUM_HEADS * KTOK * HEAD_DIM * 2);
    __bf16* Kbf   = (__bf16*)alloc((size_t)BATCH * NUM_HEADS * HW * HEAD_DIM * 2);
    __bf16* Vtb   = (__bf16*)alloc((size_t)BATCH * NUM_HEADS * HEAD_DIM * HW * 2);
    __bf16* ctxbf = (__bf16*)alloc((size_t)BATCH * KTOK * DMODEL * 2);
    float*  xbuf  = (float*) alloc((size_t)BATCH * KTOK * DMODEL * 4);

    // 1) layout/precision conversion
    transpose_hidden_kernel<<<dim3(HW / 32, DMODEL / 32, BATCH), dim3(32, 8), 0, stream>>>(
        hidden, hidbf);
    conv_bf16_kernel<<<(BATCH * KTOK * DMODEL) / 256, 256, 0, stream>>>(
        aux, auxbf, BATCH * KTOK * DMODEL);
    convT512_kernel<<<(DMODEL * DMODEL) / 256, 256, 0, stream>>>(Wq, WqT);
    convT512_kernel<<<(DMODEL * DMODEL) / 256, 256, 0, stream>>>(Wk, WkT);
    convT512_kernel<<<(DMODEL * DMODEL) / 256, 256, 0, stream>>>(Wv, WvT);
    convT512_kernel<<<(DMODEL * DMODEL) / 256, 256, 0, stream>>>(Wo, WoT);

    // 2) projections (WMMA)
    const float scale = 0.125f;  // 1/sqrt(HEAD_DIM=64), folded into Q
    gemm512_bf16<<<(BATCH * HW) / 16, 256, 0, stream>>>(
        hidbf, WkT, bk, nullptr, Kbf, nullptr, 0, 1.0f);
    gemm512_bf16<<<(BATCH * HW) / 16, 256, 0, stream>>>(
        hidbf, WvT, bv, nullptr, Vtb, nullptr, 1, 1.0f);
    gemm512_bf16<<<(BATCH * KTOK) / 16, 256, 0, stream>>>(
        auxbf, WqT, bq, nullptr, Qbf, nullptr, 2, scale);

    // 3) attention with full score strip resident in LDS
    float* aux_out  = (float*)d_out;
    float* attn_out = aux_out + (size_t)BATCH * KTOK * DMODEL;
    const size_t smem = (size_t)(16 * HW + 16 * 64 + 256 + 16) * sizeof(float);
    (void)hipFuncSetAttribute(reinterpret_cast<const void*>(attention_kernel),
                              hipFuncAttributeMaxDynamicSharedMemorySize, (int)smem);
    attention_kernel<<<dim3(KTOK / 16, NUM_HEADS, BATCH), 256, smem, stream>>>(
        Qbf, Kbf, Vtb, ctxbf, attn_out);

    // 4) output projection + residual (WMMA), then LayerNorm
    gemm512_bf16<<<(BATCH * KTOK) / 16, 256, 0, stream>>>(
        ctxbf, WoT, bo, aux, nullptr, xbuf, 3, 1.0f);
    layernorm_kernel<<<BATCH * KTOK, 256, 0, stream>>>(xbuf, ln_g, ln_b, aux_out);
}
